// SparseLookupFFN_20547123544594
// MI455X (gfx1250) — matrix operand order repaired and checked by reading, not compiled, not run on hardware
//
#include <hip/hip_runtime.h>
#include <hip/hip_bf16.h>

// ---------------------------------------------------------------------------
// SparseLookupFFN fused kernel for gfx1250 (MI455X, wave32, WMMA)
// 32 tokens / block: each bf16 B fragment feeds TWO v_wmma (acc0/acc1),
// halving L2 weight traffic and doubling wmma-per-vmem in the hot loop.
// ---------------------------------------------------------------------------

typedef __attribute__((ext_vector_type(16))) __bf16 v16bf;
typedef __attribute__((ext_vector_type(8)))  float  v8f;

#define D_MODEL   1024
#define CH        256
#define NUM_TILES 64
#define TPC       8
#define NCL       8
#define GRIDSZ    16
#define ROWS      336           // 64 sigs + 8 cluster sigs + 8 zero pad + 256 W1
#define NT_N      21            // ROWS / 16
#define KB        32            // D_MODEL / 32
#define M_TOK     32            // tokens per workgroup (two 16-row M tiles)
#define NTHREADS  256           // 8 waves
#define HSTR      261           // padded stride for h (conflict-free for 32 rows)
#define SSTR      81            // padded stride for scores
#define N_TOK     (8 * 4096)

// dynamic LDS carve (bytes)
#define A_BYTES   (2 * 16 * D_MODEL * 2)      // 65536: two 16-row bf16 A-frag sets
#define H_BYTES   (M_TOK * HSTR * 4)          // 33408
#define S_BYTES   (M_TOK * SSTR * 4)          // 10368
#define SMEM_BYTES (A_BYTES + H_BYTES + S_BYTES + M_TOK * 4 + M_TOK * 4)

__device__ __forceinline__ unsigned short f2bf(float f) {
    unsigned int u = __float_as_uint(f);
    unsigned int r = u + 0x7FFFu + ((u >> 16) & 1u);   // round-to-nearest-even
    return (unsigned short)(r >> 16);
}

__device__ __forceinline__ float signf(float v) {
    return (v > 0.f) ? 1.f : ((v < 0.f) ? -1.f : 0.f);
}

// ---------------------------------------------------------------------------
// Pack combined weight matrix (336 x 1024) into bf16 WMMA B-fragments.
// Fragment f = ((nt*KB + kb)*32 + lane); element e of lane:
//   N = nt*16 + (lane & 15);  K = kb*32 + (lane >> 4)*8 + (e & 7) + (e >> 3)*16
// ---------------------------------------------------------------------------
__global__ void pack_weights_kernel(const float* __restrict__ dirs,
                                    const float* __restrict__ W1,
                                    unsigned short* __restrict__ pb) {
    int idx = blockIdx.x * blockDim.x + threadIdx.x;
    const int total = NT_N * KB * 32 * 16;
    if (idx >= total) return;
    int e    = idx & 15;
    int lane = (idx >> 4) & 31;
    int kb   = (idx >> 9) & 31;
    int nt   = idx >> 14;
    int g    = lane >> 4;
    int n    = nt * 16 + (lane & 15);
    int k    = kb * 32 + g * 8 + (e & 7) + ((e >> 3) << 4);
    float v = 0.f;
    if (n < NUM_TILES) {
        v = signf(dirs[n * D_MODEL + k]);                 // sigs
    } else if (n < NUM_TILES + NCL) {
        int c = n - NUM_TILES;                            // cluster sigs
        float s = 0.f;
        for (int t = 0; t < TPC; ++t)
            s += signf(dirs[(c * TPC + t) * D_MODEL + k]);
        v = signf(s);
    } else if (n >= 80) {
        v = W1[(n - 80) * D_MODEL + k];                   // W1 rows
    }
    pb[idx] = f2bf(v);
}

__global__ void zero_counts_kernel(float* __restrict__ counts) {
    if (threadIdx.x < NUM_TILES) counts[threadIdx.x] = 0.f;
}

__global__ void finalize_kernel(const float* __restrict__ counts,
                                float* __restrict__ bal) {
    if (threadIdx.x == 0) {
        const float ideal = (float)N_TOK / (float)NUM_TILES;   // 512
        float acc = 0.f;
        for (int i = 0; i < NUM_TILES; ++i) {
            float d = counts[i] - ideal;
            acc += d * d;
        }
        acc /= (float)NUM_TILES;
        bal[0] = acc / (ideal * ideal + 1e-8f) * 0.01f;
    }
}

// ---------------------------------------------------------------------------
// Fused: LN -> WMMA (scores + W1) -> routing + head + spline -> output
// 256 threads (8 wave32), 32 tokens per block.
// ---------------------------------------------------------------------------
__global__ __launch_bounds__(NTHREADS)
void sparse_lookup_ffn_kernel(const float* __restrict__ x,
                              const float* __restrict__ gamma,
                              const float* __restrict__ beta,
                              const float* __restrict__ b1,
                              const float* __restrict__ W2,
                              const float* __restrict__ b2,
                              const float* __restrict__ spline_coeffs,
                              const float* __restrict__ spline_scales,
                              const float* __restrict__ dirs,
                              const float* __restrict__ output_scale,
                              const unsigned short* __restrict__ packedB,
                              float* __restrict__ out,
                              float* __restrict__ tile_out,
                              float* __restrict__ comp_out,
                              float* __restrict__ counts) {
    extern __shared__ __align__(32) char smem[];
    unsigned short* aLds = (unsigned short*)smem;                 // A fragments
    float* hLds    = (float*)(smem + A_BYTES);                    // gelu(h)
    float* sLds    = (float*)(smem + A_BYTES + H_BYTES);          // scores
    float* coefLds = (float*)(smem + A_BYTES + H_BYTES + S_BYTES);
    int*   tileLds = (int*)(smem + A_BYTES + H_BYTES + S_BYTES + M_TOK * 4);

    const int tid = threadIdx.x;
    const int r   = tid >> 3;           // token row 0..31
    const int j   = tid & 7;            // 128-column slab within row
    const int mt  = r >> 4;             // which 16-row M tile
    const int rm  = r & 15;             // row within M tile
    const long tokBase = (long)blockIdx.x * M_TOK;

    // ---- Phase 0: load x (registers), LayerNorm, write bf16 A fragments ----
    const float4* xv = (const float4*)(x + (tokBase + r) * D_MODEL + j * 128);
    float4 buf[32];
    float s = 0.f, ss = 0.f;
#pragma unroll
    for (int i = 0; i < 32; ++i) {
        float4 v = xv[i];
        buf[i] = v;
        s  += (v.x + v.y) + (v.z + v.w);
        ss += v.x * v.x + v.y * v.y + v.z * v.z + v.w * v.w;
    }
    // reduce across the 8 lanes that share one token row (contiguous in wave)
    s  += __shfl_xor(s, 1, 8);  ss += __shfl_xor(ss, 1, 8);
    s  += __shfl_xor(s, 2, 8);  ss += __shfl_xor(ss, 2, 8);
    s  += __shfl_xor(s, 4, 8);  ss += __shfl_xor(ss, 4, 8);
    const float mu   = s * (1.f / (float)D_MODEL);
    const float var  = ss * (1.f / (float)D_MODEL) - mu * mu;
    const float rstd = rsqrtf(var + 1e-5f);

    const float4* g4  = (const float4*)gamma;
    const float4* be4 = (const float4*)beta;
    const int aBase = mt * (16 * D_MODEL);     // ushort offset of this M tile
#pragma unroll
    for (int i = 0; i < 32; ++i) {
        const int k0 = j * 128 + i * 4;
        float4 v  = buf[i];
        float4 gm = g4[j * 32 + i];
        float4 bt = be4[j * 32 + i];
        float xn[4];
        xn[0] = (v.x - mu) * rstd * gm.x + bt.x;
        xn[1] = (v.y - mu) * rstd * gm.y + bt.y;
        xn[2] = (v.z - mu) * rstd * gm.z + bt.z;
        xn[3] = (v.w - mu) * rstd * gm.w + bt.w;
#pragma unroll
        for (int c = 0; c < 4; ++c) {
            const int k  = k0 + c;
            const int kb = k >> 5;
            const int w  = k & 31;
            const int g  = (w >> 3) & 1;                 // half-wave group
            const int e  = (w & 7) + ((w & 16) >> 1);    // A-frag element (ISA layout)
            aLds[aBase + kb * 512 + (rm + 16 * g) * 16 + e] = f2bf(xn[c]);
        }
    }
    __syncthreads();

    // ---- Phase 1: WMMA bf16 GEMM, each B fragment reused for both M tiles ----
    const int wave = tid >> 5;
    const int lane = tid & 31;
    const int lmod = lane & 15;
    const int mhi  = 8 * (lane >> 4);
    for (int nt = wave; nt < NT_N; nt += 8) {
        v8f acc0 = {};
        v8f acc1 = {};
#pragma unroll 2
        for (int kb = 0; kb < KB; ++kb) {
            v16bf b  = ((const v16bf*)packedB)[(nt * KB + kb) * 32 + lane];
            v16bf a0 = *(const v16bf*)(&aLds[kb * 512 + lane * 16]);
            v16bf a1 = *(const v16bf*)(&aLds[16 * D_MODEL + kb * 512 + lane * 16]);
            acc0 = __builtin_amdgcn_wmma_f32_16x16x32_bf16(
                false, a0, false, b, (short)0, acc0, false, false);
            acc1 = __builtin_amdgcn_wmma_f32_16x16x32_bf16(
                false, a1, false, b, (short)0, acc1, false, false);
        }
        if (nt < 5) {
            const int col = nt * 16 + lmod;               // 0..79
#pragma unroll
            for (int m8 = 0; m8 < 8; ++m8) {
                sLds[(m8 + mhi) * SSTR + col]        = acc0[m8];
                sLds[(16 + m8 + mhi) * SSTR + col]   = acc1[m8];
            }
        } else {
            const int col  = (nt - 5) * 16 + lmod;        // 0..255
            const float bi = b1[col];
#pragma unroll
            for (int m8 = 0; m8 < 8; ++m8) {
                float v0 = acc0[m8] + bi;
                float v1 = acc1[m8] + bi;
                hLds[(m8 + mhi) * HSTR + col] =
                    0.5f * v0 * (1.f + erff(v0 * 0.7071067811865475f));
                hLds[(16 + m8 + mhi) * HSTR + col] =
                    0.5f * v1 * (1.f + erff(v1 * 0.7071067811865475f));
            }
        }
    }
    __syncthreads();

    // ---- Phase 2: routing argmax, 256->2 tanh head, quantized spline -------
    if (tid < M_TOK) {
        const int rr = tid;
        // cluster argmax (first max, like jnp.argmax)
        float best = sLds[rr * SSTR + 64]; int ci = 0;
        for (int c = 1; c < NCL; ++c) {
            float v = sLds[rr * SSTR + 64 + c];
            if (v > best) { best = v; ci = c; }
        }
        float bl = sLds[rr * SSTR + ci * TPC]; int li = 0;
        for (int t = 1; t < TPC; ++t) {
            float v = sLds[rr * SSTR + ci * TPC + t];
            if (v > bl) { bl = v; li = t; }
        }
        const int tile = ci * TPC + li;

        float sa = b2[0], sb = b2[1];
        for (int n = 0; n < CH; ++n) {
            float h = hLds[rr * HSTR + n];
            sa += h * W2[n];
            sb += h * W2[CH + n];
        }
        const float a = tanhf(sa);
        const float b = tanhf(sb);

        int ia = (int)((a + 1.f) * 0.5f * (float)GRIDSZ);
        int ib = (int)((b + 1.f) * 0.5f * (float)GRIDSZ);
        ia = ia < 0 ? 0 : (ia > GRIDSZ - 1 ? GRIDSZ - 1 : ia);
        ib = ib < 0 ? 0 : (ib > GRIDSZ - 1 ? GRIDSZ - 1 : ib);
        const float* cc = spline_coeffs + ((tile * GRIDSZ + ia) * GRIDSZ + ib) * 3;
        const float q0 = (cc[0] > 0.3f) ? 1.f : ((cc[0] < -0.3f) ? -1.f : 0.f);
        const float q1 = (cc[1] > 0.3f) ? 1.f : ((cc[1] < -0.3f) ? -1.f : 0.f);
        const float q2 = (cc[2] > 0.3f) ? 1.f : ((cc[2] < -0.3f) ? -1.f : 0.f);
        const float la = (a + 1.f) * 8.f - (float)ia;   // cell_size = 0.125
        const float lb = (b + 1.f) * 8.f - (float)ib;
        const float spline_val = q0 + q1 * la + q2 * lb;
        const float coef = spline_val * spline_scales[tile] * output_scale[0];

        const long tok = tokBase + rr;
        comp_out[tok * 2 + 0] = a;
        comp_out[tok * 2 + 1] = b;
        tile_out[tok] = (float)tile;
        atomicAdd(&counts[tile], 1.f);
        coefLds[rr] = coef;
        tileLds[rr] = tile;
    }
    __syncthreads();

    // ---- Phase 3: out = x + coef * directions[tile] -------------------------
    const float coef = coefLds[r];
    const int   tile = tileLds[r];
    const float4* dv = (const float4*)(dirs + tile * D_MODEL + j * 128);
    float4* ov = (float4*)(out + (tokBase + r) * D_MODEL + j * 128);
#pragma unroll 4
    for (int i = 0; i < 32; ++i) {
        float4 xv4 = buf[i];
        float4 d4  = dv[i];
        float4 o;
        o.x = xv4.x + coef * d4.x;
        o.y = xv4.y + coef * d4.y;
        o.z = xv4.z + coef * d4.z;
        o.w = xv4.w + coef * d4.w;
        ov[i] = o;
    }
}

// ---------------------------------------------------------------------------
extern "C" void kernel_launch(void* const* d_in, const int* in_sizes, int n_in,
                              void* d_out, int out_size, void* d_ws, size_t ws_size,
                              hipStream_t stream) {
    (void)in_sizes; (void)n_in; (void)out_size; (void)ws_size;
    const float* x      = (const float*)d_in[0];
    const float* gamma  = (const float*)d_in[1];
    const float* beta   = (const float*)d_in[2];
    const float* W1     = (const float*)d_in[3];
    const float* b1     = (const float*)d_in[4];
    const float* W2     = (const float*)d_in[5];
    const float* b2     = (const float*)d_in[6];
    const float* spline = (const float*)d_in[7];
    const float* sscale = (const float*)d_in[8];
    const float* dirs   = (const float*)d_in[9];
    const float* oscale = (const float*)d_in[10];

    const int packed_elems = NT_N * KB * 32 * 16;           // 344064 bf16
    unsigned short* packedB = (unsigned short*)d_ws;
    float* counts = (float*)((char*)d_ws + (size_t)packed_elems * sizeof(unsigned short));

    float* out      = (float*)d_out;
    float* tile_out = out + (size_t)N_TOK * D_MODEL;
    float* comp_out = tile_out + N_TOK;
    float* bal      = comp_out + (size_t)N_TOK * 2;

    zero_counts_kernel<<<1, 64, 0, stream>>>(counts);
    pack_weights_kernel<<<(packed_elems + 255) / 256, 256, 0, stream>>>(dirs, W1, packedB);
    sparse_lookup_ffn_kernel<<<N_TOK / M_TOK, NTHREADS, SMEM_BYTES, stream>>>(
        x, gamma, beta, b1, W2, b2, spline, sscale, dirs, oscale,
        packedB, out, tile_out, comp_out, counts);
    finalize_kernel<<<1, 64, 0, stream>>>(counts, bal);
}
